// HealpyCNN_71820443124037
// MI455X (gfx1250) — matrix operand order repaired and checked by reading, not compiled, not run on hardware
//
#include <hip/hip_runtime.h>

// ---------------------------------------------------------------------------
// HealpyCNN forward for MI455X (gfx1250).
// Channel-mixing matmuls -> v_wmma_f32_16x16x32_bf16 (bf16 A/B, fp32 acc).
// Weight panels staged to LDS with global_load_async_to_lds_b128 (ASYNCcnt),
// double buffered; consumed via ds_load_b128.  Chebyshev recurrence stays in
// fp32 VALU (float4 gathers).  Working set fits 192MB L2 -> compute bound.
// GEMM invariants: M % 128 == 0 (true for every call), Nc % 16 == 0, packed
// weights zero-padded to Kpad % 32 == 0 and Npad % 64 == 0.
// ---------------------------------------------------------------------------

typedef __bf16 bf16_t;
typedef bf16_t v16bf __attribute__((ext_vector_type(16)));
typedef bf16_t v8bf  __attribute__((ext_vector_type(8)));
typedef float  v8f   __attribute__((ext_vector_type(8)));

#define BATCH 4

enum { FL_ACC = 1, FL_RELU = 2, FL_RES = 4 };

// --------------------------- Chebyshev step --------------------------------
__global__ void cheb_step_kernel(const float* __restrict__ u,
                                 const float* __restrict__ tprev,
                                 const int* __restrict__ nbr, int nnbr,
                                 float* __restrict__ out,
                                 int N, int F, int mode)
{
    long long idx = (long long)blockIdx.x * blockDim.x + threadIdx.x;
    long long total = (long long)BATCH * N * F;
    if (idx >= total) return;
    int f = (int)(idx % F);
    long long bn = idx / F;
    int n = (int)(bn % N);
    int b = (int)(bn / N);
    const float* ub = u + (long long)b * N * F;
    const int*   nb = nbr + (long long)n * nnbr;
    float s = 0.f;
    for (int j = 0; j < nnbr; ++j)
        s += ub[(long long)nb[j] * F + f];
    float Lu = u[idx] - s / (float)nnbr;
    out[idx] = (mode == 0) ? Lu : (2.0f * Lu - tprev[idx]);
}

// float4-per-thread variant for F % 4 == 0 (4x fewer gather ops).
__global__ void cheb_step4_kernel(const float4* __restrict__ u,
                                  const float4* __restrict__ tprev,
                                  const int* __restrict__ nbr, int nnbr,
                                  float4* __restrict__ out,
                                  int N, int F4, int mode)
{
    long long idx = (long long)blockIdx.x * blockDim.x + threadIdx.x;
    long long total = (long long)BATCH * N * F4;
    if (idx >= total) return;
    int f = (int)(idx % F4);
    long long bn = idx / F4;
    int n = (int)(bn % N);
    int b = (int)(bn / N);
    const float4* ub = u + (long long)b * N * F4;
    const int*    nb = nbr + (long long)n * nnbr;
    float sx = 0.f, sy = 0.f, sz = 0.f, sw = 0.f;
    for (int j = 0; j < nnbr; ++j) {
        float4 q = ub[(long long)nb[j] * F4 + f];
        sx += q.x; sy += q.y; sz += q.z; sw += q.w;
    }
    float inv = 1.0f / (float)nnbr;
    float4 uv = u[idx];
    float4 r;
    r.x = uv.x - sx * inv; r.y = uv.y - sy * inv;
    r.z = uv.z - sz * inv; r.w = uv.w - sw * inv;
    if (mode != 0) {
        float4 tp = tprev[idx];
        r.x = 2.f * r.x - tp.x; r.y = 2.f * r.y - tp.y;
        r.z = 2.f * r.z - tp.z; r.w = 2.f * r.w - tp.w;
    }
    out[idx] = r;
}

// --------------------------- weight packing --------------------------------
// W fp32 [Kc, Nc] -> Wp bf16 [npad, kpad]  (Wp[n*kpad + k]); zero padded so
// the GEMM's async panel staging needs no guards (npad % 64 == 0).
__global__ void pack_w_kernel(const float* __restrict__ W, bf16_t* __restrict__ Wp,
                              int Kc, int Nc, int kpad, int npad)
{
    int idx = blockIdx.x * blockDim.x + threadIdx.x;
    if (idx >= kpad * npad) return;
    int k = idx % kpad;
    int n = idx / kpad;
    float v = (k < Kc && n < Nc) ? W[(long long)k * Nc + n] : 0.f;
    Wp[idx] = (bf16_t)v;
}

// --------------------------- WMMA GEMM -------------------------------------
// C[m, col_off + n] (+)= A[m, :Kc] @ W[:Kc, n]  (M % 128 == 0)
// Block: 256 threads = 8 wave32, 4(M)x2(N) -> 128x64 C tile.
// Wave: 32x32 sub-tile = 2 A-frags x 2 B-frags -> 4 WMMA per 32-deep k-step.
// B panel (64 cols x 32 k, 4KB bf16) double-buffered in LDS via
// global_load_async_to_lds_b128; each of the 256 threads stages one b128.
__global__ __launch_bounds__(256) void gemm_wmma_kernel(
    const float* __restrict__ A, int lda,
    const bf16_t* __restrict__ Wp, int kpad,
    const float* __restrict__ bias,
    float* __restrict__ C, int ldc, int col_off,
    const float* __restrict__ R, int ldr,
    int Kc, int Nc, int flags)
{
    __shared__ bf16_t lb[2][64 * 32];

    const int tid  = threadIdx.x;
    const int wave = tid >> 5;
    const int lane = tid & 31;
    const int wm = wave & 3;
    const int wn = wave >> 2;
    const int row0 = blockIdx.x * 128 + wm * 32;
    const int cb0  = blockIdx.y * 64;
    const int n0   = cb0 + wn * 32;
    const bool act0 = n0 < Nc;                  // wave-uniform
    const bool act1 = (n0 + 16) < Nc;           // wave-uniform

    const int  lh  = lane & 15;
    const bool hi  = lane >= 16;
    const int  col0 = n0 + lh;
    const int  col1 = col0 + 16;

    // ---- async staging of the first B panel (k = 0..31) ----
    const int sc  = tid >> 2;                   // panel column 0..63
    const int seg = tid & 3;                    // 8-element k segment
    const bf16_t* wsrc = Wp + (long long)(cb0 + sc) * kpad + seg * 8;  // npad%64==0
    {
        unsigned la = (unsigned)(unsigned long long)&lb[0][sc * 32 + seg * 8];
        asm volatile("global_load_async_to_lds_b128 %0, %1, off"
                     :: "v"(la), "v"(wsrc) : "memory");
        asm volatile("s_wait_asynccnt 0x0" ::: "memory");
    }
    __syncthreads();

    v8f c00 = {}, c10 = {}, c01 = {}, c11 = {};
    if (act0) {
        if (flags & FL_ACC) {
#pragma unroll
            for (int i = 0; i < 8; ++i) {
                long long m0 = row0 + i + (hi ? 8 : 0);
                long long m1 = m0 + 16;
                c00[i] = C[m0 * ldc + col_off + col0];
                c10[i] = C[m1 * ldc + col_off + col0];
                if (act1) {
                    c01[i] = C[m0 * ldc + col_off + col1];
                    c11[i] = C[m1 * ldc + col_off + col1];
                }
            }
        } else {
            float b0 = bias ? bias[col0] : 0.f;
            float b1 = (bias && act1) ? bias[col1] : 0.f;
#pragma unroll
            for (int i = 0; i < 8; ++i) { c00[i] = b0; c10[i] = b0; c01[i] = b1; c11[i] = b1; }
        }
    }

    const float* pA0 = A + (long long)(row0 + lh) * lda;
    const float* pA1 = A + (long long)(row0 + 16 + lh) * lda;
    const bool vec = ((Kc & 31) == 0);          // then lda == Kc -> 16B aligned rows

    for (int k0 = 0; k0 < Kc; k0 += 32) {
        const int  buf  = (k0 >> 5) & 1;
        const bool more = (k0 + 32) < Kc;
        // kick off next panel into the other buffer while we compute
        if (more) {
            unsigned la = (unsigned)(unsigned long long)&lb[buf ^ 1][sc * 32 + seg * 8];
            asm volatile("global_load_async_to_lds_b128 %0, %1, off"
                         :: "v"(la), "v"(wsrc + k0 + 32) : "memory");
        }
        if (act0) {
            const int kbA = k0 + (hi ? 8 : 0);  // ISA 16-bit A-fragment K pattern
            v16bf a0, a1;
            if (vec) {
                float4 q0 = *(const float4*)(pA0 + kbA);
                float4 q1 = *(const float4*)(pA0 + kbA + 4);
                float4 q2 = *(const float4*)(pA0 + kbA + 16);
                float4 q3 = *(const float4*)(pA0 + kbA + 20);
                a0[0]=(bf16_t)q0.x; a0[1]=(bf16_t)q0.y; a0[2]=(bf16_t)q0.z; a0[3]=(bf16_t)q0.w;
                a0[4]=(bf16_t)q1.x; a0[5]=(bf16_t)q1.y; a0[6]=(bf16_t)q1.z; a0[7]=(bf16_t)q1.w;
                a0[8]=(bf16_t)q2.x; a0[9]=(bf16_t)q2.y; a0[10]=(bf16_t)q2.z; a0[11]=(bf16_t)q2.w;
                a0[12]=(bf16_t)q3.x; a0[13]=(bf16_t)q3.y; a0[14]=(bf16_t)q3.z; a0[15]=(bf16_t)q3.w;
                q0 = *(const float4*)(pA1 + kbA);
                q1 = *(const float4*)(pA1 + kbA + 4);
                q2 = *(const float4*)(pA1 + kbA + 16);
                q3 = *(const float4*)(pA1 + kbA + 20);
                a1[0]=(bf16_t)q0.x; a1[1]=(bf16_t)q0.y; a1[2]=(bf16_t)q0.z; a1[3]=(bf16_t)q0.w;
                a1[4]=(bf16_t)q1.x; a1[5]=(bf16_t)q1.y; a1[6]=(bf16_t)q1.z; a1[7]=(bf16_t)q1.w;
                a1[8]=(bf16_t)q2.x; a1[9]=(bf16_t)q2.y; a1[10]=(bf16_t)q2.z; a1[11]=(bf16_t)q2.w;
                a1[12]=(bf16_t)q3.x; a1[13]=(bf16_t)q3.y; a1[14]=(bf16_t)q3.z; a1[15]=(bf16_t)q3.w;
                if (more) {                     // pull next A k-slice toward L0/L2
                    __builtin_prefetch(pA0 + kbA + 32, 0, 3);
                    __builtin_prefetch(pA1 + kbA + 32, 0, 3);
                }
            } else {
                // small-K tail (Fin in {1,16}): clamped index + select, branch-free
#pragma unroll
                for (int i = 0; i < 16; ++i) {
                    int k  = kbA + i + ((i & 8) ? 8 : 0);
                    int kk = (k < Kc) ? k : 0;
                    float f0 = pA0[kk];
                    float f1 = pA1[kk];
                    a0[i] = (bf16_t)((k < Kc) ? f0 : 0.f);
                    a1[i] = (bf16_t)((k < Kc) ? f1 : 0.f);
                }
            }
            // B fragments from LDS: two 8-bf16 runs, 16 k apart = 2 x ds_load_b128
            const int cB = wn * 32 + lh;        // panel column of col0
            const int kseg = hi ? 8 : 0;        // B: lanes<16 K0..7,16..23, lanes>=16 K8..15,24..31
            v16bf bw0;
            {
                v8bf lo = *(const v8bf*)&lb[buf][cB * 32 + kseg];
                v8bf hv = *(const v8bf*)&lb[buf][cB * 32 + kseg + 16];
#pragma unroll
                for (int i = 0; i < 8; ++i) { bw0[i] = lo[i]; bw0[i + 8] = hv[i]; }
            }
            c00 = __builtin_amdgcn_wmma_f32_16x16x32_bf16(false, a0, false, bw0,
                                                          (short)0, c00, false, false);
            c10 = __builtin_amdgcn_wmma_f32_16x16x32_bf16(false, a1, false, bw0,
                                                          (short)0, c10, false, false);
            if (act1) {
                v16bf bw1;
                v8bf lo = *(const v8bf*)&lb[buf][(cB + 16) * 32 + kseg];
                v8bf hv = *(const v8bf*)&lb[buf][(cB + 16) * 32 + kseg + 16];
#pragma unroll
                for (int i = 0; i < 8; ++i) { bw1[i] = lo[i]; bw1[i + 8] = hv[i]; }
                c01 = __builtin_amdgcn_wmma_f32_16x16x32_bf16(false, a0, false, bw1,
                                                              (short)0, c01, false, false);
                c11 = __builtin_amdgcn_wmma_f32_16x16x32_bf16(false, a1, false, bw1,
                                                              (short)0, c11, false, false);
            }
        }
        if (more) {
            asm volatile("s_wait_asynccnt 0x0" ::: "memory");
            __syncthreads();
        }
    }

    if (act0) {
#pragma unroll
        for (int i = 0; i < 8; ++i) {
            long long m0 = row0 + i + (hi ? 8 : 0);
            long long m1 = m0 + 16;
            float v00 = c00[i], v10 = c10[i], v01 = c01[i], v11 = c11[i];
            if (flags & FL_RELU) {
                v00 = fmaxf(v00, 0.f); v10 = fmaxf(v10, 0.f);
                v01 = fmaxf(v01, 0.f); v11 = fmaxf(v11, 0.f);
            }
            if (flags & FL_RES) {
                v00 += R[m0 * ldr + col0];
                v10 += R[m1 * ldr + col0];
                if (act1) { v01 += R[m0 * ldr + col1]; v11 += R[m1 * ldr + col1]; }
            }
            C[m0 * ldc + col_off + col0] = v00;
            C[m1 * ldc + col_off + col0] = v10;
            if (act1) {
                C[m0 * ldc + col_off + col1] = v01;
                C[m1 * ldc + col_off + col1] = v11;
            }
        }
    }
}

// ------------------------- pool4 + relu + slice ----------------------------
__global__ void pool4_relu_kernel(const float* __restrict__ in, float* __restrict__ out,
                                  int Nout, int F, int ldo, int col_off)
{
    long long idx = (long long)blockIdx.x * blockDim.x + threadIdx.x;
    long long total = (long long)BATCH * Nout * F;
    if (idx >= total) return;
    int f = (int)(idx % F);
    long long bp = idx / F;
    int p = (int)(bp % Nout);
    int b = (int)(bp / Nout);
    const float* ip = in + ((long long)b * (4 * Nout) + 4 * (long long)p) * F + f;
    float s = 0.25f * (ip[0] + ip[F] + ip[2 * F] + ip[3 * F]);
    s = fmaxf(s, 0.f);
    out[((long long)b * Nout + p) * ldo + col_off + f] = s;
}

// --------------------- global mean pool + linear head ----------------------
__global__ void head_kernel(const float* __restrict__ h, const float* __restrict__ W,
                            const float* __restrict__ bias, float* __restrict__ out,
                            int N, int F, int C)
{
    int b = blockIdx.x / C;
    int c = blockIdx.x % C;
    __shared__ float red[256];
    float acc = 0.f;
    for (int f = threadIdx.x; f < F; f += blockDim.x) {
        float s = 0.f;
        const float* hp = h + (long long)b * N * F + f;
        for (int n = 0; n < N; ++n) s += hp[(long long)n * F];
        acc += (s / (float)N) * W[f * C + c];
    }
    red[threadIdx.x] = acc;
    __syncthreads();
    for (int st = 128; st > 0; st >>= 1) {
        if ((int)threadIdx.x < st) red[threadIdx.x] += red[threadIdx.x + st];
        __syncthreads();
    }
    if (threadIdx.x == 0) out[b * C + c] = red[0] + bias[c];
}

// ===========================================================================
// Host-side orchestration
// ===========================================================================
static inline int cdiv(int a, int b) { return (a + b - 1) / b; }

struct ConvP { const float* b; const float* w; };

static void pack_w_launch(hipStream_t s, const float* W, bf16_t* wp,
                          int Kc, int Nc, int& kpad, int& npad)
{
    kpad = (Kc + 31) & ~31;
    npad = (Nc + 63) & ~63;      // async panel staging reads cols 0..npad-1
    int tot = kpad * npad;
    pack_w_kernel<<<cdiv(tot, 256), 256, 0, s>>>(W, wp, Kc, Nc, kpad, npad);
}

static void launch_gemm(hipStream_t s, const float* A, int lda,
                        const bf16_t* wp, int kpad, const float* bias,
                        float* C, int ldc, int coff,
                        const float* R, int ldr,
                        int M, int Kc, int Nc, int flags)
{
    dim3 g(M / 128, cdiv(Nc, 64));              // M % 128 == 0 for every call
    gemm_wmma_kernel<<<g, 256, 0, s>>>(A, lda, wp, kpad, bias, C, ldc, coff,
                                       R, ldr, Kc, Nc, flags);
}

static void launch_cheb(hipStream_t s, const float* u, const float* tprev,
                        const int* nbr, int nnbr, float* out, int N, int F, int mode)
{
    if ((F & 3) == 0) {
        int F4 = F >> 2;
        int tot = BATCH * N * F4;
        cheb_step4_kernel<<<cdiv(tot, 256), 256, 0, s>>>(
            (const float4*)u, (const float4*)tprev, nbr, nnbr, (float4*)out, N, F4, mode);
    } else {
        int tot = BATCH * N * F;
        cheb_step_kernel<<<cdiv(tot, 256), 256, 0, s>>>(u, tprev, nbr, nnbr, out, N, F, mode);
    }
}

// One Chebyshev graph conv: out = concat(T0..T_{K-1}) @ W + b  (+epilogue).
static void cheb_conv(hipStream_t s, bf16_t* wp,
                      const float* v, const int* nbr, int nnbr, int K,
                      int Fin, int Fout, const float* W, const float* bias,
                      float* out, int ldc, int coff, int N,
                      float* tA, float* tB,
                      int final_flags, const float* R, int ldr)
{
    const int M = BATCH * N;
    int kpad, npad;
    // k = 0 : T0 = v
    pack_w_launch(s, W, wp, Fin, Fout, kpad, npad);
    launch_gemm(s, v, Fin, wp, kpad, bias, out, ldc, coff, R, ldr,
                M, Fin, Fout, (K == 1) ? final_flags : 0);
    if (K == 1) return;

    // k = 1 : T1 = L(v)
    launch_cheb(s, v, nullptr, nbr, nnbr, tA, N, Fin, 0);
    pack_w_launch(s, W + (long long)Fin * Fout, wp, Fin, Fout, kpad, npad);
    launch_gemm(s, tA, Fin, wp, kpad, nullptr, out, ldc, coff, R, ldr,
                M, Fin, Fout, FL_ACC | (K == 2 ? final_flags : 0));

    const float* tprev = v;
    float* tcur = tA;
    for (int k = 2; k < K; ++k) {
        float* tnext = (tprev == v) ? tB : (float*)tprev;   // elementwise-safe reuse
        launch_cheb(s, tcur, tprev, nbr, nnbr, tnext, N, Fin, 1);
        pack_w_launch(s, W + (long long)k * Fin * Fout, wp, Fin, Fout, kpad, npad);
        launch_gemm(s, tnext, Fin, wp, kpad, nullptr, out, ldc, coff, R, ldr,
                    M, Fin, Fout, FL_ACC | (k == K - 1 ? final_flags : 0));
        tprev = tcur; tcur = tnext;
    }
}

static void pool4(hipStream_t s, const float* in, float* out,
                  int Nout, int F, int ldo, int coff)
{
    int tot = BATCH * Nout * F;
    pool4_relu_kernel<<<cdiv(tot, 256), 256, 0, s>>>(in, out, Nout, F, ldo, coff);
}

static void wide_block(hipStream_t s, bf16_t* wp, const float* v, int N, int Fin,
                       int Fmid, const ConvP c[7], const int* i8, const int* i20,
                       float* hout, float* b1, float* b2, float* tA, float* tB)
{
    const int F2 = 2 * Fmid, ldh = 4 * Fmid, Np = N / 4;
    // branch1: K4 -> K8 -> K2 -> pool
    cheb_conv(s, wp, v,  i8, 8, 4, Fin,  Fmid, c[0].w, c[0].b, b1, Fmid, 0, N, tA, tB, FL_RELU, nullptr, 0);
    cheb_conv(s, wp, b1, i8, 8, 8, Fmid, Fmid, c[1].w, c[1].b, b2, Fmid, 0, N, tA, tB, FL_RELU, nullptr, 0);
    cheb_conv(s, wp, b2, i8, 8, 2, Fmid, Fmid, c[2].w, c[2].b, b1, Fmid, 0, N, tA, tB, 0, nullptr, 0);
    pool4(s, b1, hout, Np, Fmid, ldh, 0);
    // branch2: K8 -> K2 -> pool
    cheb_conv(s, wp, v,  i8, 8, 8, Fin,  Fmid, c[3].w, c[3].b, b1, Fmid, 0, N, tA, tB, FL_RELU, nullptr, 0);
    cheb_conv(s, wp, b1, i8, 8, 2, Fmid, Fmid, c[4].w, c[4].b, b2, Fmid, 0, N, tA, tB, 0, nullptr, 0);
    pool4(s, b2, hout, Np, Fmid, ldh, Fmid);
    // branch3: K12 (20-nbr) -> K2 -> pool
    cheb_conv(s, wp, v,  i20, 20, 12, Fin, F2, c[5].w, c[5].b, b1, F2, 0, N, tA, tB, FL_RELU, nullptr, 0);
    cheb_conv(s, wp, b1, i8,  8,  2,  F2,  F2, c[6].w, c[6].b, b2, F2, 0, N, tA, tB, 0, nullptr, 0);
    pool4(s, b2, hout, Np, F2, ldh, 2 * Fmid);
}

static void res_block(hipStream_t s, bf16_t* wp, const float* v, int N, int F,
                      ConvP a, ConvP b, const int* i8, const int* i20,
                      float* tmp, float* out, float* tA, float* tB)
{
    cheb_conv(s, wp, v,   i8,  8,  6, F, F, a.w, a.b, tmp, F, 0, N, tA, tB, FL_RELU, nullptr, 0);
    cheb_conv(s, wp, tmp, i20, 20, 10, F, F, b.w, b.b, out, F, 0, N, tA, tB, FL_RELU | FL_RES, v, F);
}

static void mid_block(hipStream_t s, bf16_t* wp, const float* v, int N, int F,
                      ConvP c, ConvP pp, const int* i8,
                      float* tmp, float* out, float* tA, float* tB)
{
    const int Fo = 2 * F;
    cheb_conv(s, wp, v, i8, 8, 8, F, Fo, c.w, c.b, tmp, Fo, 0, N, tA, tB, FL_RELU, nullptr, 0);
    // pseudo conv p=1: [B,N,Fo] viewed as [B*N/4, 4*Fo] -> GEMM + relu
    int kpad, npad;
    pack_w_launch(s, pp.w, wp, 4 * Fo, Fo, kpad, npad);
    launch_gemm(s, tmp, 4 * Fo, wp, kpad, pp.b, out, Fo, 0, nullptr, 0,
                BATCH * N / 4, 4 * Fo, Fo, FL_RELU);
}

extern "C" void kernel_launch(void* const* d_in, const int* in_sizes, int n_in,
                              void* d_out, int out_size, void* d_ws, size_t ws_size,
                              hipStream_t stream)
{
    (void)in_sizes; (void)n_in; (void)out_size; (void)ws_size;
    // ---- inputs (JAX pytree flattening: dict keys sorted alphabetically) ----
    const float* x = (const float*)d_in[0];
    int p = 1;
    auto F32 = [&](int i) { return (const float*)d_in[i]; };
    ConvP fin_a, fin_b;                                     // 'final'
    fin_a.b = F32(p++); fin_a.w = F32(p++);
    fin_b.b = F32(p++); fin_b.w = F32(p++);
    const float* head_b = F32(p++); const float* head_w = F32(p++);   // 'head'
    ConvP m1c, m1p, m2c, m2p;                               // 'mid1','mid2'
    m1c.b = F32(p++); m1c.w = F32(p++); m1p.b = F32(p++); m1p.w = F32(p++);
    m2c.b = F32(p++); m2c.w = F32(p++); m2p.b = F32(p++); m2p.w = F32(p++);
    ConvP r2a[3], r2b[3], r3a[3], r3b[3], r4a[3], r4b[3];   // 'res2','res3','res4'
    for (int t = 0; t < 3; ++t) { r2a[t].b = F32(p++); r2a[t].w = F32(p++);
                                  r2b[t].b = F32(p++); r2b[t].w = F32(p++); }
    for (int t = 0; t < 3; ++t) { r3a[t].b = F32(p++); r3a[t].w = F32(p++);
                                  r3b[t].b = F32(p++); r3b[t].w = F32(p++); }
    for (int t = 0; t < 3; ++t) { r4a[t].b = F32(p++); r4a[t].w = F32(p++);
                                  r4b[t].b = F32(p++); r4b[t].w = F32(p++); }
    ConvP wb1c[7], wb2c[7];                                 // 'wb1','wb2'
    for (int t = 0; t < 7; ++t) { wb1c[t].b = F32(p++); wb1c[t].w = F32(p++); }
    for (int t = 0; t < 7; ++t) { wb2c[t].b = F32(p++); wb2c[t].w = F32(p++); }
    const int *n8[5], *n20[5];
    for (int r = 0; r < 5; ++r) { n8[r] = (const int*)d_in[p++]; n20[r] = (const int*)d_in[p++]; }

    // ---- workspace layout (static offsets into d_ws) ----
    const long long BIG = 25165824;   // 786432 rows * 32 ch  (100.7 MB fp32)
    const long long ACT = 12582912;   // 196608 rows * 64 ch  (50.3 MB fp32)
    float* bb1 = (float*)d_ws;
    float* bb2 = bb1 + BIG;
    float* tA  = bb2 + BIG;
    float* tB  = tA + BIG;
    float* hA  = tB + BIG;
    float* hB  = hA + ACT;
    bf16_t* wp = (bf16_t*)(hB + ACT);  // packed-weight staging (<= 2M elems)

    hipStream_t s = stream;

    // ---- forward pass ----
    wide_block(s, wp, x,  196608, 1,  16, wb1c, n8[0], n20[0], hA, bb1, bb2, tA, tB); // -> r1, 64ch
    wide_block(s, wp, hA, 49152, 64, 32, wb2c, n8[1], n20[1], hB, bb1, bb2, tA, tB);  // -> r2, 128ch

    const float* cur = hB; float* nxt = hA;
    for (int t = 0; t < 3; ++t) {                                     // res2 @ r2, 128ch
        res_block(s, wp, cur, 12288, 128, r2a[t], r2b[t], n8[2], n20[2], bb1, nxt, tA, tB);
        cur = nxt; nxt = (nxt == hA) ? hB : hA;
    }
    mid_block(s, wp, cur, 12288, 128, m1c, m1p, n8[2], bb1, nxt, tA, tB);  // -> r3, 256ch
    cur = nxt; nxt = (nxt == hA) ? hB : hA;
    for (int t = 0; t < 3; ++t) {                                     // res3 @ r3, 256ch
        res_block(s, wp, cur, 3072, 256, r3a[t], r3b[t], n8[3], n20[3], bb1, nxt, tA, tB);
        cur = nxt; nxt = (nxt == hA) ? hB : hA;
    }
    mid_block(s, wp, cur, 3072, 256, m2c, m2p, n8[3], bb1, nxt, tA, tB);   // -> r4, 512ch
    cur = nxt; nxt = (nxt == hA) ? hB : hA;
    for (int t = 0; t < 3; ++t) {                                     // res4 @ r4, 512ch
        res_block(s, wp, cur, 768, 512, r4a[t], r4b[t], n8[4], n20[4], bb1, nxt, tA, tB);
        cur = nxt; nxt = (nxt == hA) ? hB : hA;
    }
    // final: K8 (8-nbr) then K12 (20-nbr), both 512->512 @ r4
    cheb_conv(s, wp, cur, n8[4],  8,  8, 512, 512, fin_a.w, fin_a.b, bb1, 512, 0, 768, tA, tB, FL_RELU, nullptr, 0);
    cheb_conv(s, wp, bb1, n20[4], 20, 12, 512, 512, fin_b.w, fin_b.b, nxt, 512, 0, 768, tA, tB, FL_RELU, nullptr, 0);

    // global mean over pixels + linear head -> (4, 3)
    head_kernel<<<BATCH * 3, 256, 0, s>>>(nxt, head_w, head_b, (float*)d_out, 768, 512, 3);
}
